// TRLMasked_33191507263799
// MI455X (gfx1250) — matrix operand order repaired and checked by reading, not compiled, not run on hardware
//
#include <hip/hip_runtime.h>
#include <hip/hip_bf16.h>
#include <stdint.h>

// ---------------------------------------------------------------------------
// Tensor-ring layer for MI455X (gfx1250, wave32).
//   Stage 1: build effective weight in workspace, emitted directly as bf16.
//   Stage 1b: convert x fp32 -> bf16 once.
//   Stage 2: out = x @ W + bias via v_wmma_f32_16x16x32_bf16 with double-
//            buffered LDS staged by GLOBAL_LOAD_ASYNC_TO_LDS_B128 (ASYNCcnt).
// ---------------------------------------------------------------------------

typedef __attribute__((ext_vector_type(16))) __bf16 v16bf;
typedef __attribute__((ext_vector_type(8)))  float  v8f;
typedef __attribute__((ext_vector_type(4)))  int    v4i;

#define NIN   4096   // 8^4 input features
#define OF_   1024   // output features
#define LDT   72     // LDS row stride (halfwords): 144B rows, 16B aligned

#if __has_builtin(__builtin_amdgcn_global_load_async_to_lds_b128) && \
    __has_builtin(__builtin_amdgcn_s_wait_asynccnt)
#define USE_ASYNC 1
#else
#define USE_ASYNC 0
#endif

typedef __attribute__((address_space(1))) v4i* gptr_v4i;
typedef __attribute__((address_space(3))) v4i* lptr_v4i;

__device__ __forceinline__ unsigned pack2bf(float lo, float hi) {
  // round-to-nearest-even fp32 -> bf16, packed pair
  unsigned ul = __float_as_uint(lo);
  unsigned uh = __float_as_uint(hi);
  ul = (ul + 0x7FFFu + ((ul >> 16) & 1u)) >> 16;
  uh = (uh + 0x7FFFu + ((uh >> 16) & 1u)) >> 16;
  return (ul & 0xFFFFu) | (uh << 16);
}

__device__ __forceinline__ void async_copy_b128(const void* g, void* l) {
#if USE_ASYNC
  __builtin_amdgcn_global_load_async_to_lds_b128(
      (gptr_v4i)(uintptr_t)g,
      (lptr_v4i)(unsigned)(uintptr_t)l,
      0, 0);
#else
  *(uint4*)l = *(const uint4*)g;   // sync fallback: global_load_b128 + ds_store_b128
#endif
}

__device__ __forceinline__ void wait_async_6() {
#if USE_ASYNC
  __builtin_amdgcn_s_wait_asynccnt(6);
#endif
}
__device__ __forceinline__ void wait_async_0() {
#if USE_ASYNC
  __builtin_amdgcn_s_wait_asynccnt(0);
#endif
}

// ---------------- Stage 1: tensor-ring weight construction ----------------
// t1[(i,a,b),(q,s,t,o)] = sum_p f0[i,a,b,p] * core[p,q,s,t,o]   (512 x 32768)
__global__ __launch_bounds__(256) void k_t1(const float* __restrict__ f0,
                                            const float* __restrict__ core,
                                            float* __restrict__ t1) {
  unsigned e = blockIdx.x * 256u + threadIdx.x;   // 16,777,216 elements
  unsigned c = e & 32767u;
  unsigned r = e >> 15;
  float acc = 0.f;
#pragma unroll
  for (int p = 0; p < 8; ++p)
    acc += f0[r * 8u + p] * core[(unsigned)p * 32768u + c];
  t1[e] = acc;
}

// t2[(i,a),(j,c),(s,t,o)] = sum_{b,q} f1[j,b,c,q] * t1[(i,a,b),(q,s,t,o)]
__global__ __launch_bounds__(256) void k_t2(const float* __restrict__ f1,
                                            const float* __restrict__ t1,
                                            float* __restrict__ t2) {
  unsigned e = blockIdx.x * 256u + threadIdx.x;   // 16,777,216 elements
  unsigned sto = e & 4095u;
  unsigned jc  = (e >> 12) & 63u;
  unsigned ia  = e >> 18;
  unsigned j = jc >> 3, c = jc & 7u;
  float acc = 0.f;
  for (int b = 0; b < 8; ++b) {
    const float* t1p = t1 + (((size_t)(ia * 8u + b)) << 15) + sto;
    const float* f1p = f1 + ((j * 8u + b) * 8u + c) * 8u;
#pragma unroll
    for (int q = 0; q < 8; ++q)
      acc += f1p[q] * t1p[(unsigned)q * 4096u];
  }
  t2[e] = acc;
}

// t3[(i,a,j),(k,d),(t,o)] = sum_{c,s} f2[k,c,d,s] * t2[(i,a),(j,c),(s,t,o)]
__global__ __launch_bounds__(256) void k_t3(const float* __restrict__ f2,
                                            const float* __restrict__ t2,
                                            float* __restrict__ t3) {
  unsigned e = blockIdx.x * 256u + threadIdx.x;   // 16,777,216 elements
  unsigned to_ = e & 511u;
  unsigned kd  = (e >> 9) & 63u;
  unsigned iaj = e >> 15;
  unsigned k = kd >> 3, d = kd & 7u;
  unsigned ia = iaj >> 3, j = iaj & 7u;
  float acc = 0.f;
  for (int c = 0; c < 8; ++c) {
    const float* f2p = f2 + ((k * 8u + c) * 8u + d) * 8u;
    const float* t2p = t2 + (size_t)(ia * 64u + j * 8u + c) * 4096u + to_;
#pragma unroll
    for (int s = 0; s < 8; ++s)
      acc += f2p[s] * t2p[(unsigned)s * 512u];
  }
  t3[e] = acc;
}

// t4[i,j,k,l,o] = sum_{a,d,t} f3[l,d,a,t] * t3[(i,a,j),(k,d),(t,o)]  (ring trace)
__global__ __launch_bounds__(256) void k_t4(const float* __restrict__ f3,
                                            const float* __restrict__ t3,
                                            float* __restrict__ t4) {
  unsigned e = blockIdx.x * 256u + threadIdx.x;   // 262,144 elements
  unsigned o = e & 63u;
  unsigned ijkl = e >> 6;
  unsigned l = ijkl & 7u;
  unsigned k = (ijkl >> 3) & 7u;
  unsigned j = (ijkl >> 6) & 7u;
  unsigned i = ijkl >> 9;
  float acc = 0.f;
  for (int a = 0; a < 8; ++a) {
    for (int d = 0; d < 8; ++d) {
      const float* f3p = f3 + ((l * 8u + d) * 8u + a) * 8u;
      const float* t3p = t3 + (((size_t)((i * 64u + a * 8u + j) * 64u + k * 8u + d)) << 9) + o;
#pragma unroll
      for (int t = 0; t < 8; ++t)
        acc += f3p[t] * t3p[(unsigned)t * 64u];
    }
  }
  t4[e] = acc;
}

// Wb[n,f] (bf16, packed pairs) = sum_o t4[n,o] * out_factor[o,f]
__global__ __launch_bounds__(256) void k_w(const float* __restrict__ t4,
                                           const float* __restrict__ of_,
                                           unsigned* __restrict__ Wb) {
  unsigned e = blockIdx.x * 256u + threadIdx.x;   // 2,097,152 packed pairs
  unsigned fp = e & 511u;       // pair index within row
  unsigned n  = e >> 9;
  unsigned f0i = fp * 2u;
  float a0 = 0.f, a1 = 0.f;
#pragma unroll 8
  for (int o = 0; o < 64; ++o) {
    float tv = t4[n * 64u + o];
    a0 += tv * of_[(unsigned)o * 1024u + f0i];
    a1 += tv * of_[(unsigned)o * 1024u + f0i + 1u];
  }
  Wb[e] = pack2bf(a0, a1);      // row n, cols f0i..f0i+1
}

// x fp32 -> bf16 (packed), 8 floats per thread
__global__ __launch_bounds__(256) void k_xc(const float* __restrict__ X,
                                            uint4* __restrict__ Xb) {
  unsigned e = blockIdx.x * 256u + threadIdx.x;   // 2,097,152 uint4 outputs
  const float4* xp = (const float4*)X + (size_t)e * 2u;
  float4 lo = xp[0], hi = xp[1];
  uint4 r;
  r.x = pack2bf(lo.x, lo.y); r.y = pack2bf(lo.z, lo.w);
  r.z = pack2bf(hi.x, hi.y); r.w = pack2bf(hi.z, hi.w);
  Xb[e] = r;
}

// ---------------- Stage 2: out = x @ W + bias (bf16 WMMA) ----------------
// Block: 256 threads (8 waves), tile 128(M) x 64(N) x 64(K) per stage.
// Wave -> 32x32 patch: 2x2 accumulators, 8 WMMAs per K-stage.
// Double-buffered LDS fed by async global->LDS b128 copies (6 per thread/stage).
__global__ __launch_bounds__(256) void k_gemm(const unsigned short* __restrict__ Xb,
                                              const unsigned short* __restrict__ Wb,
                                              const float* __restrict__ bias,
                                              float* __restrict__ out) {
  __shared__ unsigned short As[2][128 * LDT];  // 2 x 18432 B
  __shared__ unsigned short Bs[2][64 * LDT];   // 2 x  9216 B

  const int tid  = threadIdx.x;
  const int bm   = blockIdx.x * 128;
  const int bn   = blockIdx.y * 64;
  const int wave = tid >> 5, lane = tid & 31;
  const int waveM = wave & 3, waveN = wave >> 2;   // 4 x 2 wave grid
  const int hl   = lane >> 4;                      // half-wave select
  const int mrow = lane & 15;

  v8f zero = {0.f, 0.f, 0.f, 0.f, 0.f, 0.f, 0.f, 0.f};
  v8f acc[2][2];
  acc[0][0] = zero; acc[0][1] = zero; acc[1][0] = zero; acc[1][1] = zero;

  // per-thread async-copy chunk mapping (16B chunks, 8 bf16 each)
  // A tile: 128 rows x 64 cols -> 1024 chunks (4/thread)
  // B tile:  64 rows x 64 cols ->  512 chunks (2/thread)
  int aRow[4], aSeg[4], bRow[2], bSeg[2];
#pragma unroll
  for (int c = 0; c < 4; ++c) { int ch = tid + c * 256; aRow[c] = ch >> 3; aSeg[c] = ch & 7; }
#pragma unroll
  for (int c = 0; c < 2; ++c) { int ch = tid + c * 256; bRow[c] = ch >> 3; bSeg[c] = ch & 7; }

#define ISSUE_TILE(buf, k0)                                                              \
  do {                                                                                   \
    _Pragma("unroll")                                                                    \
    for (int c = 0; c < 4; ++c)                                                          \
      async_copy_b128(Xb + (size_t)(bm + aRow[c]) * NIN + (k0) + aSeg[c] * 8,            \
                      &As[buf][aRow[c] * LDT + aSeg[c] * 8]);                            \
    _Pragma("unroll")                                                                    \
    for (int c = 0; c < 2; ++c)                                                          \
      async_copy_b128(Wb + (size_t)((k0) + bRow[c]) * OF_ + bn + bSeg[c] * 8,            \
                      &Bs[buf][bRow[c] * LDT + bSeg[c] * 8]);                            \
  } while (0)

  ISSUE_TILE(0, 0);

  const int NT = NIN / 64;   // 64 K-stages
  for (int kt = 0; kt < NT; ++kt) {
    const int buf = kt & 1;
    if (kt + 1 < NT) {
      ISSUE_TILE(buf ^ 1, (kt + 1) * 64);
      wait_async_6();          // oldest 6 (this stage's tile) complete, newest 6 in flight
    } else {
      wait_async_0();
    }
    __syncthreads();           // all waves' copies for this stage visible

    // two K=32 phases per stage
#pragma unroll
    for (int kp = 0; kp < 2; ++kp) {
      union { v16bf v; unsigned u[8]; } afr[2], bfr[2];
      // A fragments: 16-bit A 16x32 layout — m=lane%16, half-wave K offset 8,
      // VGPR v holds K pair {2v,2v+1} (v<4) / {16+2(v-4),+1} (v>=4)
#pragma unroll
      for (int sm = 0; sm < 2; ++sm) {
        const int base = (waveM * 32 + sm * 16 + mrow) * LDT + kp * 32;
#pragma unroll
        for (int v = 0; v < 8; ++v) {
          const int kb = (v < 4 ? 2 * v : 8 + 2 * v) + hl * 8;
          afr[sm].u[v] = *(const unsigned*)&As[buf][base + kb];
        }
      }
      // B fragments: 16-bit B 32x16 layout — n=lane%16, K order same as A
#pragma unroll
      for (int sn = 0; sn < 2; ++sn) {
        const int n0 = waveN * 32 + sn * 16;
        const int cb = kp * 32 * LDT + n0 + mrow;
#pragma unroll
        for (int v = 0; v < 8; ++v) {
          const int kb = (v < 4 ? 2 * v : 8 + 2 * v) + hl * 8;
          bfr[sn].u[v] = (unsigned)Bs[buf][cb + kb * LDT] | ((unsigned)Bs[buf][cb + (kb + 1) * LDT] << 16);
        }
      }
#pragma unroll
      for (int sm = 0; sm < 2; ++sm)
#pragma unroll
        for (int sn = 0; sn < 2; ++sn)
          acc[sm][sn] = __builtin_amdgcn_wmma_f32_16x16x32_bf16(
              false, afr[sm].v, false, bfr[sn].v,
              (short)0, acc[sm][sn], false, false);
    }
    __syncthreads();           // this stage's buffer free before it is re-filled
  }
#undef ISSUE_TILE

  // Epilogue: C/D layout — VGPR r: M = r + 8*halfwave, N = lane%16
#pragma unroll
  for (int sm = 0; sm < 2; ++sm) {
#pragma unroll
    for (int sn = 0; sn < 2; ++sn) {
      const int N = bn + waveN * 32 + sn * 16 + mrow;
      const float bv = bias[N];
#pragma unroll
      for (int r = 0; r < 8; ++r) {
        const int M = bm + waveM * 32 + sm * 16 + hl * 8 + r;
        out[(size_t)M * OF_ + N] = acc[sm][sn][r] + bv;
      }
    }
  }
}

// ---------------------------------------------------------------------------
extern "C" void kernel_launch(void* const* d_in, const int* in_sizes, int n_in,
                              void* d_out, int out_size, void* d_ws, size_t ws_size,
                              hipStream_t stream) {
  (void)in_sizes; (void)n_in; (void)out_size; (void)ws_size;
  const float* x    = (const float*)d_in[0];
  const float* f0   = (const float*)d_in[1];
  const float* f1   = (const float*)d_in[2];
  const float* f2   = (const float*)d_in[3];
  const float* f3   = (const float*)d_in[4];
  const float* core = (const float*)d_in[5];
  const float* ofac = (const float*)d_in[6];
  const float* bias = (const float*)d_in[7];
  float* out = (float*)d_out;

  char* ws = (char*)d_ws;                               // layout (bytes):
  float*          t1 = (float*)(ws);                    // [0,   64MB) t1 / reused as t3
  float*          t2 = (float*)(ws + ((size_t)64  << 20));  // [64, 128MB) t2
  float*          t4 = (float*)(ws + ((size_t)128 << 20));  // [128,129MB) t4
  unsigned short* Wb = (unsigned short*)(ws + ((size_t)129 << 20)); // [129,137MB) W bf16
  unsigned short* Xb = (unsigned short*)(ws + ((size_t)137 << 20)); // [137,169MB) x bf16
  float* t3 = t1;

  k_xc<<<8192,  256, 0, stream>>>(x, (uint4*)Xb);
  k_t1<<<65536, 256, 0, stream>>>(f0, core, t1);
  k_t2<<<65536, 256, 0, stream>>>(f1, t1, t2);
  k_t3<<<65536, 256, 0, stream>>>(f2, t2, t3);
  k_t4<<<1024,  256, 0, stream>>>(f3, t3, t4);
  k_w <<<8192,  256, 0, stream>>>(t4, ofac, (unsigned*)Wb);

  dim3 grid(NIN / 128, OF_ / 64);                       // 32 x 16 blocks
  k_gemm<<<grid, 256, 0, stream>>>(Xb, Wb, bias, out);
}
